// memory_60833916780935
// MI455X (gfx1250) — matrix operand order repaired and checked by reading, not compiled, not run on hardware
//
#include <hip/hip_runtime.h>
#include <hip/hip_bf16.h>
#include <math.h>

typedef __attribute__((ext_vector_type(16))) _Float16 v16h;
typedef __attribute__((ext_vector_type(8)))  _Float16 v8h;
typedef __attribute__((ext_vector_type(8)))  float    v8f;

#define LN_EPS 1e-6f

// ---------------------------------------------------------------------------
// WMMA fragment loaders (wave32, 16x16x32 f16 shapes per CDNA5 ISA layouts)
// A 16x32 (MxK): lanes 0-15: M=lane, halves 0-7 -> K 0-7, halves 8-15 -> K 16-23
//                lanes 16-31: M=lane-16, halves 0-7 -> K 8-15, 8-15 -> K 24-31
// B 32x16 (KxN): lanes 0-15: N=lane,   halves 0-7 -> K 0-7, halves 8-15 -> K 16-23
//                lanes 16-31: N=lane-16, halves 0-7 -> K 8-15, 8-15 -> K 24-31
// C/D 16x16 f32: elem r: lanes 0-15 -> M=r, lanes 16-31 -> M=8+r; N=lane&15
// ---------------------------------------------------------------------------

__device__ __forceinline__ v16h ld_a_f16(const _Float16* __restrict__ src, int ld, int koff) {
  const int lane = threadIdx.x & 31;
  const _Float16* r = src + (lane & 15) * ld;
  const int kb = koff + ((lane >> 4) << 3);
  v16h a;
#pragma unroll
  for (int i = 0; i < 8; ++i) { a[i] = r[kb + i]; a[i + 8] = r[kb + 16 + i]; }
  return a;
}

__device__ __forceinline__ v16h ld_a_f32cvt(const float* __restrict__ src, int ld, int koff) {
  const int lane = threadIdx.x & 31;
  const float* r = src + (lane & 15) * ld;
  const int kb = koff + ((lane >> 4) << 3);
  v16h a;
#pragma unroll
  for (int i = 0; i < 8; ++i) { a[i] = (_Float16)r[kb + i]; a[i + 8] = (_Float16)r[kb + 16 + i]; }
  return a;
}

// B[k][n] = src[n*ld + k]  (rows of src indexed by N, contiguous along K)
// -> two runs of 8 consecutive halves per lane (K kb..kb+7, kb+16..kb+23): 2x global_load_b128
__device__ __forceinline__ v16h ld_bT_f16(const _Float16* __restrict__ src, int ld, int koff) {
  const int lane = threadIdx.x & 31;
  const _Float16* r = src + (lane & 15) * ld;
  const int kb = koff + ((lane >> 4) << 3);
  v16h b;
#pragma unroll
  for (int i = 0; i < 8; ++i) { b[i] = r[kb + i]; b[i + 8] = r[kb + 16 + i]; }
  return b;
}

// B[k][n] = src[k*ld + n0+n], f32 source converted (weight matrices)
__device__ __forceinline__ v16h ld_b_f32cvt(const float* __restrict__ src, int ld, int koff, int n0) {
  const int lane = threadIdx.x & 31;
  const int n = n0 + (lane & 15);
  const int kb = koff + ((lane >> 4) << 3);
  v16h b;
#pragma unroll
  for (int i = 0; i < 8; ++i) { b[i] = (_Float16)src[(kb + i) * ld + n]; b[i + 8] = (_Float16)src[(kb + 16 + i) * ld + n]; }
  return b;
}

// ---------------------------------------------------------------------------
// K1: logits = xt @ wb_w + b; weight1 = softmax over classes (->ws);
//     weight2 = softmax over HW; class_feat1 = weight2 @ xt (->ws)
// one block per batch, 320 threads
// ---------------------------------------------------------------------------
__global__ void __launch_bounds__(320) k1_pool(const float* __restrict__ x,
                                               const float* __restrict__ wbw,
                                               const float* __restrict__ wbb,
                                               float* __restrict__ w1,
                                               float* __restrict__ cf1) {
  const int b = blockIdx.x;
  const int t = threadIdx.x;
  __shared__ float logits[1024 * 5];
  __shared__ float red[320 * 5];
  __shared__ float mxs[5], sms[5];
  const float* xb = x + b * 64 * 1024;

  for (int p = t; p < 1024; p += 320) {
    float acc[5];
#pragma unroll
    for (int n = 0; n < 5; ++n) acc[n] = wbb[n];
    for (int c = 0; c < 64; ++c) {
      const float xv = xb[c * 1024 + p];
#pragma unroll
      for (int n = 0; n < 5; ++n) acc[n] += xv * wbw[c * 5 + n];
    }
#pragma unroll
    for (int n = 0; n < 5; ++n) logits[p * 5 + n] = acc[n];
  }
  __syncthreads();

  // weight1: softmax over classes
  float* w1b = w1 + b * 1024 * 5;
  for (int p = t; p < 1024; p += 320) {
    float m = -1e30f;
#pragma unroll
    for (int n = 0; n < 5; ++n) m = fmaxf(m, logits[p * 5 + n]);
    float e[5], s = 0.f;
#pragma unroll
    for (int n = 0; n < 5; ++n) { e[n] = __expf(logits[p * 5 + n] - m); s += e[n]; }
    const float inv = 1.f / s;
#pragma unroll
    for (int n = 0; n < 5; ++n) w1b[p * 5 + n] = e[n] * inv;
  }

  // weight2 reductions over HW
  {
    float lm[5];
#pragma unroll
    for (int n = 0; n < 5; ++n) lm[n] = -1e30f;
    for (int p = t; p < 1024; p += 320)
#pragma unroll
      for (int n = 0; n < 5; ++n) lm[n] = fmaxf(lm[n], logits[p * 5 + n]);
#pragma unroll
    for (int n = 0; n < 5; ++n) red[t * 5 + n] = lm[n];
  }
  __syncthreads();
  if (t < 5) {
    float m = -1e30f;
    for (int i = 0; i < 320; ++i) m = fmaxf(m, red[i * 5 + t]);
    mxs[t] = m;
  }
  __syncthreads();
  {
    float ls[5];
#pragma unroll
    for (int n = 0; n < 5; ++n) ls[n] = 0.f;
    for (int p = t; p < 1024; p += 320)
#pragma unroll
      for (int n = 0; n < 5; ++n) ls[n] += __expf(logits[p * 5 + n] - mxs[n]);
#pragma unroll
    for (int n = 0; n < 5; ++n) red[t * 5 + n] = ls[n];
  }
  __syncthreads();
  if (t < 5) {
    float s = 0.f;
    for (int i = 0; i < 320; ++i) s += red[i * 5 + t];
    sms[t] = s;
  }
  __syncthreads();

  // class_feat1[b][n][c], t = n*64 + c  (exactly 320 outputs)
  {
    const int n = t >> 6, c = t & 63;
    const float mx = mxs[n], inv = 1.f / sms[n];
    float acc = 0.f;
    for (int p = 0; p < 1024; ++p)
      acc += __expf(logits[p * 5 + n] - mx) * inv * xb[c * 1024 + p];
    cf1[(b * 5 + n) * 64 + c] = acc;
  }
}

// ---------------------------------------------------------------------------
// K2: MHA1 (n_head=1, Lq=5, Lk=5, d_k=96, d_v=64) + post-LN, one block/batch
// ---------------------------------------------------------------------------
__global__ void __launch_bounds__(128) k2_mha1(const float* __restrict__ cf1,
                                               const float* __restrict__ mem,
                                               const float* __restrict__ wq, const float* __restrict__ bq,
                                               const float* __restrict__ wk, const float* __restrict__ bk,
                                               const float* __restrict__ wv, const float* __restrict__ bv,
                                               const float* __restrict__ fc, const float* __restrict__ fcb,
                                               const float* __restrict__ lng, const float* __restrict__ lnb,
                                               float* __restrict__ cfa) {
  const int b = blockIdx.x;
  const int t = threadIdx.x;
  __shared__ float qh[5 * 96], kh[5 * 96], vh[5 * 64], sc[5 * 6], y[5 * 64];
  const float* q = cf1 + b * 5 * 64;

  for (int idx = t; idx < 5 * 96; idx += 128) {
    const int n = idx / 96, j = idx % 96;
    float aq = bq[j], ak = bk[j];
    for (int i = 0; i < 64; ++i) aq += q[n * 64 + i] * wq[i * 96 + j];
    for (int i = 0; i < 96; ++i) ak += mem[n * 96 + i] * wk[i * 96 + j];
    qh[idx] = aq; kh[idx] = ak;
  }
  for (int idx = t; idx < 5 * 64; idx += 128) {
    const int n = idx >> 6, j = idx & 63;
    float av = bv[j];
    for (int i = 0; i < 96; ++i) av += mem[n * 96 + i] * wv[i * 64 + j];
    vh[idx] = av;
  }
  __syncthreads();
  if (t < 25) {
    const int n = t / 5, m2 = t % 5;
    float s = 0.f;
    for (int j = 0; j < 96; ++j) s += qh[n * 96 + j] * kh[m2 * 96 + j];
    sc[n * 6 + m2] = s * 0.10206207261596577f; // 1/sqrt(96)
  }
  __syncthreads();
  if (t < 5) {
    float m = -1e30f;
    for (int i = 0; i < 5; ++i) m = fmaxf(m, sc[t * 6 + i]);
    float s = 0.f;
    for (int i = 0; i < 5; ++i) { const float e = __expf(sc[t * 6 + i] - m); sc[t * 6 + i] = e; s += e; }
    const float inv = 1.f / s;
    for (int i = 0; i < 5; ++i) sc[t * 6 + i] *= inv;
  }
  __syncthreads();
  for (int idx = t; idx < 5 * 64; idx += 128) {
    const int n = idx >> 6, j = idx & 63;
    float o = 0.f;
    for (int m2 = 0; m2 < 5; ++m2) o += sc[n * 6 + m2] * vh[m2 * 64 + j];
    y[idx] = o;
  }
  __syncthreads();
  for (int idx = t; idx < 5 * 64; idx += 128) {
    const int n = idx >> 6, j = idx & 63;
    float acc = fcb[j] + q[n * 64 + j];   // residual
    for (int k = 0; k < 64; ++k) acc += y[n * 64 + k] * fc[k * 64 + j];
    vh[idx] = acc;                        // reuse vh as pre-LN buffer
  }
  __syncthreads();
  if (t < 5) {
    float mu = 0.f;
    for (int j = 0; j < 64; ++j) mu += vh[t * 64 + j];
    mu *= (1.f / 64.f);
    float var = 0.f;
    for (int j = 0; j < 64; ++j) { const float d = vh[t * 64 + j] - mu; var += d * d; }
    var *= (1.f / 64.f);
    sc[t * 6 + 0] = mu;
    sc[t * 6 + 1] = rsqrtf(var + LN_EPS);
  }
  __syncthreads();
  for (int idx = t; idx < 5 * 64; idx += 128) {
    const int n = idx >> 6, j = idx & 63;
    cfa[(b * 5 + n) * 64 + j] = (vh[idx] - sc[n * 6 + 0]) * sc[n * 6 + 1] * lng[j] + lnb[j];
  }
}

// ---------------------------------------------------------------------------
// K3: cf[b][p][c] = sum_n weight1[b][p][n] * cfa[b][n][c]
// ---------------------------------------------------------------------------
__global__ void __launch_bounds__(256) k3_redis(const float* __restrict__ w1,
                                                const float* __restrict__ cfa,
                                                float* __restrict__ cf) {
  const int idx = blockIdx.x * 256 + threadIdx.x;   // < 16*1024*64
  const int c = idx & 63;
  const int bp = idx >> 6;
  const int b = bp >> 10;
  const float* w = w1 + bp * 5;
  const float* s = cfa + b * 5 * 64 + c;
  float acc = 0.f;
#pragma unroll
  for (int n = 0; n < 5; ++n) acc += w[n] * s[n * 64];
  cf[idx] = acc;
}

// ---------------------------------------------------------------------------
// K4: WMMA projection GEMM: [16384x64] @ [64x256] + bias -> f16
// grid.y selects Q/K/V; 4 waves per block, 16x16 tile per wave, 2 WMMAs/tile
// Q,K stored row-major [row][256]; V stored TRANSPOSED as Vt[b][h][dv][key]
// (the 8 C rows per lane are consecutive keys -> one 16B v8h store per lane)
// ---------------------------------------------------------------------------
__global__ void __launch_bounds__(128) k4_proj(const float* __restrict__ cf,
                                               const float* __restrict__ wq, const float* __restrict__ bq,
                                               const float* __restrict__ wk, const float* __restrict__ bk,
                                               const float* __restrict__ wv, const float* __restrict__ bv,
                                               _Float16* __restrict__ Qh,
                                               _Float16* __restrict__ Kh,
                                               _Float16* __restrict__ Vt) {
  const int wave = threadIdx.x >> 5;
  const int tile = blockIdx.x * 4 + wave;   // 16384 tiles = 1024 mt x 16 nt
  const int mt = tile >> 4, nt = tile & 15;
  const int m0 = mt * 16, n0 = nt * 16;

  const float* W; const float* B;
  if (blockIdx.y == 0)      { W = wq; B = bq; }
  else if (blockIdx.y == 1) { W = wk; B = bk; }
  else                      { W = wv; B = bv; }

  const v16h a0 = ld_a_f32cvt(cf + (size_t)m0 * 64, 64, 0);
  const v16h a1 = ld_a_f32cvt(cf + (size_t)m0 * 64, 64, 32);
  const v16h b0 = ld_b_f32cvt(W, 256, 0, n0);
  const v16h b1 = ld_b_f32cvt(W, 256, 32, n0);

  v8f c = {};
  c = __builtin_amdgcn_wmma_f32_16x16x32_f16(false, a0, false, b0, (short)0, c, false, false);
  c = __builtin_amdgcn_wmma_f32_16x16x32_f16(false, a1, false, b1, (short)0, c, false, false);

  const int lane = threadIdx.x & 31;
  const int col = n0 + (lane & 15);
  const float bias = B[col];
  const int mo = (lane >> 4) << 3;

  if (blockIdx.y == 2) {
    // transposed V store: Vt[((b*4 + h)*64 + dv)*1024 + key]
    v8h pack;
#pragma unroll
    for (int r = 0; r < 8; ++r) pack[r] = (_Float16)(c[r] + bias);
    const int row0 = m0 + mo;                 // 8 consecutive keys
    _Float16* dst = Vt + (((size_t)(row0 >> 10) * 4 + (col >> 6)) * 64 + (col & 63)) * 1024
                       + (row0 & 1023);
    *(v8h*)dst = pack;
  } else {
    _Float16* O = (blockIdx.y == 0) ? Qh : Kh;
#pragma unroll
    for (int r = 0; r < 8; ++r)
      O[(size_t)(m0 + mo + r) * 256 + col] = (_Float16)(c[r] + bias);
  }
}

// ---------------------------------------------------------------------------
// K5: flash attention, one wave per (b, head, 16-row q tile)
// 32 keys per iteration: 4 WMMAs for the 16x32 score strip, chunked online
// softmax, P staged as full 16x32 LDS tile (no zero pad), 4 WMMAs for P.V
// with full K=32 contraction; V read from transposed layout (b128 loads).
// ---------------------------------------------------------------------------
__global__ void __launch_bounds__(32) k5_attn(const _Float16* __restrict__ Qh,
                                              const _Float16* __restrict__ Kh,
                                              const _Float16* __restrict__ Vt,
                                              float* __restrict__ attO) {
  const int bi = blockIdx.x;         // 16*4*64
  const int qt = bi & 63;
  const int h  = (bi >> 6) & 3;
  const int b  = bi >> 8;
  const int lane = threadIdx.x & 31;
  __shared__ _Float16 P[16 * 32];

  const _Float16* Qt = Qh + (size_t)(b * 1024 + qt * 16) * 256 + h * 64;
  const v16h qa0 = ld_a_f16(Qt, 256, 0);
  const v16h qa1 = ld_a_f16(Qt, 256, 32);

  const _Float16* Kb = Kh + (size_t)b * 1024 * 256 + h * 64;      // rows = keys
  const _Float16* Vb = Vt + ((size_t)(b * 4 + h) * 64) * 1024;    // rows = dv

  float rmax[8], rsum[8];
  v8f o0 = {}, o1 = {}, o2 = {}, o3 = {};
#pragma unroll
  for (int r = 0; r < 8; ++r) { rmax[r] = -1e30f; rsum[r] = 0.f; }

  for (int kt = 0; kt < 32; ++kt) {
    const _Float16* K0 = Kb + (size_t)(kt * 32) * 256;            // keys kt*32 .. +15
    const _Float16* K1 = K0 + 16 * 256;                           // keys +16 .. +31

    if (kt + 1 < 32) {   // prefetch next key strip (K rows + V columns)
      __builtin_prefetch(Kb + (size_t)((kt + 1) * 32 + (lane & 15)) * 256, 0, 3);
      __builtin_prefetch(Kb + (size_t)((kt + 1) * 32 + 16 + (lane & 15)) * 256, 0, 3);
      __builtin_prefetch(Vb + (size_t)((lane & 15) + 16 * (lane >> 4)) * 1024 + (kt + 1) * 32, 0, 3);
    }

    const v16h kA0 = ld_bT_f16(K0, 256, 0);
    const v16h kA1 = ld_bT_f16(K0, 256, 32);
    const v16h kB0 = ld_bT_f16(K1, 256, 0);
    const v16h kB1 = ld_bT_f16(K1, 256, 32);
    v8f slo = {}, shi = {};
    slo = __builtin_amdgcn_wmma_f32_16x16x32_f16(false, qa0, false, kA0, (short)0, slo, false, false);
    slo = __builtin_amdgcn_wmma_f32_16x16x32_f16(false, qa1, false, kA1, (short)0, slo, false, false);
    shi = __builtin_amdgcn_wmma_f32_16x16x32_f16(false, qa0, false, kB0, (short)0, shi, false, false);
    shi = __builtin_amdgcn_wmma_f32_16x16x32_f16(false, qa1, false, kB1, (short)0, shi, false, false);

    float plo[8], phi[8], corr[8];
#pragma unroll
    for (int r = 0; r < 8; ++r) {
      const float vlo = slo[r] * 0.125f;         // 1/sqrt(64)
      const float vhi = shi[r] * 0.125f;
      float tm = fmaxf(vlo, vhi);                // row max over 32-key chunk
      tm = fmaxf(tm, __shfl_xor(tm, 1, 32));
      tm = fmaxf(tm, __shfl_xor(tm, 2, 32));
      tm = fmaxf(tm, __shfl_xor(tm, 4, 32));
      tm = fmaxf(tm, __shfl_xor(tm, 8, 32));
      const float nm = fmaxf(rmax[r], tm);
      corr[r] = __expf(rmax[r] - nm);
      rmax[r] = nm;
      plo[r] = __expf(vlo - nm);
      phi[r] = __expf(vhi - nm);
      float ts = plo[r] + phi[r];                // row sum over chunk
      ts += __shfl_xor(ts, 1, 32);
      ts += __shfl_xor(ts, 2, 32);
      ts += __shfl_xor(ts, 4, 32);
      ts += __shfl_xor(ts, 8, 32);
      rsum[r] = rsum[r] * corr[r] + ts;
    }
#pragma unroll
    for (int r = 0; r < 8; ++r) { o0[r] *= corr[r]; o1[r] *= corr[r]; o2[r] *= corr[r]; o3[r] *= corr[r]; }

    const int mo = (lane >> 4) << 3;
#pragma unroll
    for (int r = 0; r < 8; ++r) {
      P[(mo + r) * 32 + (lane & 15)]      = (_Float16)plo[r];
      P[(mo + r) * 32 + 16 + (lane & 15)] = (_Float16)phi[r];
    }
    __syncthreads();

    // A fragment from full 16x32 P tile in LDS (ds loads, no padding)
    v16h pa;
    {
      const int m = lane & 15;
      const int kb = (lane >> 4) << 3;
#pragma unroll
      for (int i = 0; i < 8; ++i) { pa[i] = P[m * 32 + kb + i]; pa[i + 8] = P[m * 32 + kb + 16 + i]; }
    }

    // V in B layout directly from transposed Vt: B[k][n] = Vt[dv=n][key=k]
    const v16h vb0 = ld_bT_f16(Vb +  0 * 1024 + kt * 32, 1024, 0);
    const v16h vb1 = ld_bT_f16(Vb + 16 * 1024 + kt * 32, 1024, 0);
    const v16h vb2 = ld_bT_f16(Vb + 32 * 1024 + kt * 32, 1024, 0);
    const v16h vb3 = ld_bT_f16(Vb + 48 * 1024 + kt * 32, 1024, 0);
    o0 = __builtin_amdgcn_wmma_f32_16x16x32_f16(false, pa, false, vb0, (short)0, o0, false, false);
    o1 = __builtin_amdgcn_wmma_f32_16x16x32_f16(false, pa, false, vb1, (short)0, o1, false, false);
    o2 = __builtin_amdgcn_wmma_f32_16x16x32_f16(false, pa, false, vb2, (short)0, o2, false, false);
    o3 = __builtin_amdgcn_wmma_f32_16x16x32_f16(false, pa, false, vb3, (short)0, o3, false, false);
    __syncthreads();
  }

  const int mo = (lane >> 4) << 3;
#pragma unroll
  for (int r = 0; r < 8; ++r) {
    const float inv = 1.f / rsum[r];
    const size_t row = (size_t)(b * 1024 + qt * 16 + mo + r);
    const int col = h * 64 + (lane & 15);
    attO[row * 256 + col +  0] = o0[r] * inv;
    attO[row * 256 + col + 16] = o1[r] * inv;
    attO[row * 256 + col + 32] = o2[r] * inv;
    attO[row * 256 + col + 48] = o3[r] * inv;
  }
}

// ---------------------------------------------------------------------------
// K6: final fc (256->64) + residual + LayerNorm + transpose to [b,64,H,W]
// one block (64 threads) per (b,p) row
// ---------------------------------------------------------------------------
__global__ void __launch_bounds__(64) k6_out(const float* __restrict__ attO,
                                             const float* __restrict__ cf,
                                             const float* __restrict__ fc,
                                             const float* __restrict__ fcb,
                                             const float* __restrict__ lng,
                                             const float* __restrict__ lnb,
                                             float* __restrict__ out) {
  const int row = blockIdx.x;   // 16384
  const int c = threadIdx.x;    // 64
  __shared__ float sdata[64];
  const float* a = attO + (size_t)row * 256;
  float acc = fcb[c] + cf[(size_t)row * 64 + c];  // residual
  for (int k = 0; k < 256; ++k) acc += a[k] * fc[k * 64 + c];
  sdata[c] = acc;
  __syncthreads();
  float mu = 0.f;
  for (int j = 0; j < 64; ++j) mu += sdata[j];
  mu *= (1.f / 64.f);
  float var = 0.f;
  for (int j = 0; j < 64; ++j) { const float d = sdata[j] - mu; var += d * d; }
  var *= (1.f / 64.f);
  const float y = (acc - mu) * rsqrtf(var + LN_EPS) * lng[c] + lnb[c];
  const int b = row >> 10, p = row & 1023;
  out[(size_t)(b * 64 + c) * 1024 + p] = y;
}

// ---------------------------------------------------------------------------
extern "C" void kernel_launch(void* const* d_in, const int* in_sizes, int n_in,
                              void* d_out, int out_size, void* d_ws, size_t ws_size,
                              hipStream_t stream) {
  const float* x     = (const float*)d_in[0];
  const float* mem   = (const float*)d_in[1];
  const float* wbw   = (const float*)d_in[2];
  const float* wbb   = (const float*)d_in[3];
  const float* a1wq  = (const float*)d_in[4];
  const float* a1bq  = (const float*)d_in[5];
  const float* a1wk  = (const float*)d_in[6];
  const float* a1bk  = (const float*)d_in[7];
  const float* a1wv  = (const float*)d_in[8];
  const float* a1bv  = (const float*)d_in[9];
  const float* a1fc  = (const float*)d_in[10];
  const float* a1fcb = (const float*)d_in[11];
  const float* a1lng = (const float*)d_in[12];
  const float* a1lnb = (const float*)d_in[13];
  const float* a2wq  = (const float*)d_in[14];
  const float* a2bq  = (const float*)d_in[15];
  const float* a2wk  = (const float*)d_in[16];
  const float* a2bk  = (const float*)d_in[17];
  const float* a2wv  = (const float*)d_in[18];
  const float* a2bv  = (const float*)d_in[19];
  const float* a2fc  = (const float*)d_in[20];
  const float* a2fcb = (const float*)d_in[21];
  const float* a2lng = (const float*)d_in[22];
  const float* a2lnb = (const float*)d_in[23];

  char* ws = (char*)d_ws;
  size_t off = 0;
  auto carve = [&](size_t bytes) -> void* {
    void* p = ws + off;
    off = (off + bytes + 255) & ~(size_t)255;
    return p;
  };
  float*    w1   = (float*)carve(16 * 1024 * 5 * sizeof(float));            // weight1
  float*    cf1  = (float*)carve(16 * 5 * 64 * sizeof(float));              // class_feat1
  float*    cfa  = (float*)carve(16 * 5 * 64 * sizeof(float));              // after MHA1
  float*    cf   = (float*)carve((size_t)16 * 1024 * 64 * sizeof(float));   // redistributed / residual
  _Float16* Qh   = (_Float16*)carve((size_t)16 * 1024 * 256 * sizeof(_Float16));
  _Float16* Kh   = (_Float16*)carve((size_t)16 * 1024 * 256 * sizeof(_Float16));
  _Float16* Vt   = (_Float16*)carve((size_t)16 * 1024 * 256 * sizeof(_Float16)); // transposed V
  float*    attO = (float*)carve((size_t)16 * 1024 * 256 * sizeof(float));

  k1_pool<<<16, 320, 0, stream>>>(x, wbw, wbb, w1, cf1);
  k2_mha1<<<16, 128, 0, stream>>>(cf1, mem, a1wq, a1bq, a1wk, a1bk, a1wv, a1bv,
                                  a1fc, a1fcb, a1lng, a1lnb, cfa);
  k3_redis<<<4096, 256, 0, stream>>>(w1, cfa, cf);
  dim3 g4(4096, 3);
  k4_proj<<<g4, 128, 0, stream>>>(cf, a2wq, a2bq, a2wk, a2bk, a2wv, a2bv, Qh, Kh, Vt);
  k5_attn<<<16 * 4 * 64, 32, 0, stream>>>(Qh, Kh, Vt, attO);
  k6_out<<<16384, 64, 0, stream>>>(attO, cf, a2fc, a2fcb, a2lng, a2lnb, (float*)d_out);
}
